// GeometricProductDense_2388001817291
// MI455X (gfx1250) — matrix operand order repaired and checked by reading, not compiled, not run on hardware
//
#include <hip/hip_runtime.h>

// CDNA5 / gfx1250, wave32.  GA dense layer as 16 blade-indexed GEMMs (k = blockIdx.z):
//   out[b,u,k] = sum_{n,j} sign(j^k,j) * x[b,n,j^k] * w[u,n,j] + bias[u,k]
// Phase 1: one-shot f32->bf16 conversion of w and x into d_ws (done once, ~10 MB).
// Phase 2: bf16 GEMM; w streams via async global->LDS (no VALU), x gets a 1-xor
//          sign + XOR-permute during LDS store (k uniform per block).
// Fallback (small ws): fused kernel converting in-loop (round-2 style).

typedef __attribute__((ext_vector_type(16))) __bf16 v16bf;
typedef __attribute__((ext_vector_type(8)))  float  v8f;

#define B_DIM 128
#define NI    8192                 // GEMM K  (N_IN * 16 blades)
#define UNITS 512                  // GEMM N
#define OUTC  (UNITS * 16)
#define TM    64
#define TN    128
#define TK    32
#define NKT   (NI / TK)            // 256 K-steps

#define WELEMS ((size_t)UNITS * NI)   // 4 Mi elems
#define XELEMS ((size_t)B_DIM * NI)   // 1 Mi elems

#if __has_builtin(__builtin_amdgcn_global_load_async_to_lds_b128)
#define GA_HAS_ASYNC 1
#else
#define GA_HAS_ASYNC 0
#endif

__device__ __forceinline__ unsigned f2bf_u(float f) {
    unsigned u = __float_as_uint(f);
    u += 0x7FFFu + ((u >> 16) & 1u);
    return u >> 16;
}
__device__ __forceinline__ unsigned pack_bf16(float lo, float hi) {
    return f2bf_u(lo) | (f2bf_u(hi) << 16);
}
__device__ __forceinline__ int ga_swaps_odd(int a, int b) {
    int t = a >> 1, s = 0;
    while (t) { s += __popc(t & b); t >>= 1; }
    return s & 1;
}

union FragAB { uint4 q[2]; v16bf v; };
union AccU   { v8f v; float f[8]; };

__device__ __forceinline__ v16bf loadFragA(const unsigned short* lAbuf, int row, int hh) {
    // 16-bit A 16x32 layout: lanes 0-15 -> K{0..7,16..23}; lanes 16-31 -> K{8..15,24..31}
    FragAB f;
    const unsigned short* p = lAbuf + row * TK;
    f.q[0] = *(const uint4*)(p + hh * 8);
    f.q[1] = *(const uint4*)(p + 16 + hh * 8);
    return f.v;
}
__device__ __forceinline__ v16bf loadFragB(const unsigned short* lBbuf, int row, int hh) {
    // 16-bit B 32x16 layout: lanes 0-15 -> K{0..7,16..23}; lanes 16-31 -> K{8..15,24..31}
    FragAB f;
    const unsigned short* p = lBbuf + row * TK;
    f.q[0] = *(const uint4*)(p + hh * 8);
    f.q[1] = *(const uint4*)(p + 16 + hh * 8);
    return f.v;
}

#if GA_HAS_ASYNC
// builtin signature (from probe diagnostics): param0 is AS(1) pointer to a
// 16-byte int vector; param1 is the LDS-side pointer; then (offset, cpol).
typedef int ga_v4i __attribute__((vector_size(16)));
typedef __attribute__((address_space(1))) ga_v4i* ga_gp1;
typedef __attribute__((address_space(3))) ga_v4i* ga_lp3;

__device__ __forceinline__ void async_cp_b128(const unsigned short* g, unsigned short* l) {
    // integer casts: global generic == AS1 numerically; LDS generic low 32b == AS3 offset
    __builtin_amdgcn_global_load_async_to_lds_b128(
        (ga_gp1)(unsigned long long)g,
        (ga_lp3)(unsigned)(unsigned long long)l, 0, 0);
}
__device__ __forceinline__ void async_wait0() {
    asm volatile("s_wait_asynccnt 0x0" ::: "memory");
}
#endif

// ======================= phase 1: one-shot f32 -> bf16 =======================
__global__ __launch_bounds__(256)
void cvt_to_bf16(const float* __restrict__ w, const float* __restrict__ x,
                 unsigned short* __restrict__ wsW, unsigned short* __restrict__ wsX) {
    const size_t t    = (size_t)blockIdx.x * blockDim.x + threadIdx.x;
    const size_t base = t * 8;                     // 8 elems/thread, no straddle
    const float* src; unsigned short* dst; size_t off;
    if (base < WELEMS) { src = w; dst = wsW; off = base; }
    else               { src = x; dst = wsX; off = base - WELEMS; }
    const float4* s4 = (const float4*)(src + off);
    const float4 a = s4[0], b = s4[1];
    uint4 o;
    o.x = pack_bf16(a.x, a.y);  o.y = pack_bf16(a.z, a.w);
    o.z = pack_bf16(b.x, b.y);  o.w = pack_bf16(b.z, b.w);
    *(uint4*)(dst + off) = o;
}

// ---- A tile (bf16 source): sign-xor on packed pair + XOR-permuted b32 stores
template <bool KODD>
__device__ __forceinline__ void storeA_bf(unsigned short* base, int c0, int kbe,
                                          uint4 av, const unsigned* sm32) {
    const unsigned pr[4] = { av.x, av.y, av.z, av.w };
#pragma unroll
    for (int p = 0; p < 4; ++p) {
        unsigned u = pr[p] ^ sm32[p];
        if (KODD) u = (u >> 16) | (u << 16);       // swap halves: blade parity flip
        *(unsigned*)(base + ((c0 + 2 * p) ^ kbe)) = u;   // even offset -> 4B aligned
    }
}

// ======================= phase 2: bf16 GEMM (ws path) ========================
template <bool KODD>
__device__ __forceinline__
void ga_gemm_ws_body(const unsigned short* __restrict__ xbf,
                     const unsigned short* __restrict__ wbf,
                     const float* __restrict__ bias, float* __restrict__ out) {
    __shared__ __align__(16) unsigned short lA[2][TM * TK];   // 2 x 4 KB
    __shared__ __align__(16) unsigned short lB[2][TN * TK];   // 2 x 8 KB

    const int tid   = threadIdx.x;
    const int uBase = blockIdx.x * TN;
    const int mBase = blockIdx.y * TM;
    const int kb    = blockIdx.z;           // output blade (uniform per block)
    const int kbe   = kb & 14;

    // A loader: 8 bf16/thread (half of one 16-blade group)
    const int aRow = tid >> 2;
    const int q    = tid & 3;
    const int ah   = q >> 1;
    const int c0   = (q & 1) << 3;
    const unsigned short* aPtr = xbf + (size_t)(mBase + aRow) * NI + ah * 16 + c0;

    unsigned sgn = 0;
#pragma unroll
    for (int c = 0; c < 16; ++c)
        sgn |= (unsigned)ga_swaps_odd(c, c ^ kb) << c;
    unsigned sm32[4];
#pragma unroll
    for (int p = 0; p < 4; ++p) {
        const int c = c0 + 2 * p;
        sm32[p] = (((sgn >> c) & 1u) << 15) | (((sgn >> (c + 1)) & 1u) << 31);
    }

    // B loader: 16 bf16/thread, pure copy
    const int bRow  = tid >> 1;
    const int bHalf = tid & 1;
    const unsigned short* bPtr = wbf + (size_t)(uBase + bRow) * NI + bHalf * 16;
    unsigned short* bDst0 = &lB[0][bRow * TK + bHalf * 16];
    unsigned short* bDst1 = &lB[1][bRow * TK + bHalf * 16];

    // compute roles: 8 waves, 2(M) x 4(N), each wave a 32x32 C tile
    const int wave = tid >> 5;
    const int wr   = wave >> 2;
    const int wc   = wave & 3;
    const int lane = tid & 31;
    const int r16  = lane & 15;
    const int hh   = lane >> 4;

    v8f acc[2][2] = {};

    // prologue: stage K-tile 0
    {
#if GA_HAS_ASYNC
        async_cp_b128(bPtr,     bDst0);
        async_cp_b128(bPtr + 8, bDst0 + 8);
#else
        ((uint4*)bDst0)[0] = ((const uint4*)bPtr)[0];
        ((uint4*)bDst0)[1] = ((const uint4*)bPtr)[1];
#endif
        const uint4 av = *(const uint4*)aPtr;
        storeA_bf<KODD>(&lA[0][aRow * TK + ah * 16], c0, kbe, av, sm32);
#if GA_HAS_ASYNC
        async_wait0();
#endif
    }
    __syncthreads();

    for (int kt = 0; kt < NKT; ++kt) {
        const int cur = kt & 1, nxt = cur ^ 1;
        unsigned short* bDst = nxt ? bDst1 : bDst0;

        uint4 av;
#if !GA_HAS_ASYNC
        uint4 bv0, bv1;
#endif
        const bool pre = (kt + 1 < NKT);
        if (pre) {
            const unsigned short* bp = bPtr + (size_t)(kt + 1) * TK;
#if GA_HAS_ASYNC
            async_cp_b128(bp,     bDst);
            async_cp_b128(bp + 8, bDst + 8);
#else
            bv0 = ((const uint4*)bp)[0];
            bv1 = ((const uint4*)bp)[1];
#endif
            av = *(const uint4*)(aPtr + (size_t)(kt + 1) * TK);
        }

        // WMMA on current buffer (uniform control flow -> EXEC all ones)
        v16bf fa0 = loadFragA(lA[cur], wr * 32 + 0 * 16 + r16, hh);
        v16bf fa1 = loadFragA(lA[cur], wr * 32 + 1 * 16 + r16, hh);
        v16bf fb0 = loadFragB(lB[cur], wc * 32 + 0 * 16 + r16, hh);
        v16bf fb1 = loadFragB(lB[cur], wc * 32 + 1 * 16 + r16, hh);
        acc[0][0] = __builtin_amdgcn_wmma_f32_16x16x32_bf16(false, fa0, false, fb0,
                        (short)0, acc[0][0], false, false);
        acc[0][1] = __builtin_amdgcn_wmma_f32_16x16x32_bf16(false, fa0, false, fb1,
                        (short)0, acc[0][1], false, false);
        acc[1][0] = __builtin_amdgcn_wmma_f32_16x16x32_bf16(false, fa1, false, fb0,
                        (short)0, acc[1][0], false, false);
        acc[1][1] = __builtin_amdgcn_wmma_f32_16x16x32_bf16(false, fa1, false, fb1,
                        (short)0, acc[1][1], false, false);

        if (pre) {
            storeA_bf<KODD>(&lA[nxt][aRow * TK + ah * 16], c0, kbe, av, sm32);
#if !GA_HAS_ASYNC
            ((uint4*)bDst)[0] = bv0;
            ((uint4*)bDst)[1] = bv1;
#endif
        }
#if GA_HAS_ASYNC
        async_wait0();
#endif
        __syncthreads();
    }

#pragma unroll
    for (int nj = 0; nj < 2; ++nj) {
        const int ucol = uBase + wc * 32 + nj * 16 + r16;
        const float bv = bias[ucol * 16 + kb];
#pragma unroll
        for (int mi = 0; mi < 2; ++mi) {
            AccU a; a.v = acc[mi][nj];
            const int rowBase = mBase + wr * 32 + mi * 16 + hh * 8;
#pragma unroll
            for (int e = 0; e < 8; ++e)
                out[(size_t)(rowBase + e) * OUTC + ucol * 16 + kb] = a.f[e] + bv;
        }
    }
}

__global__ __launch_bounds__(256)
void ga_gemm_ws(const unsigned short* __restrict__ xbf, const unsigned short* __restrict__ wbf,
                const float* __restrict__ bias, float* __restrict__ out) {
    if (blockIdx.z & 1) ga_gemm_ws_body<true>(xbf, wbf, bias, out);
    else                ga_gemm_ws_body<false>(xbf, wbf, bias, out);
}

// ================= fallback: fused in-loop conversion (small ws) =============
template <bool KODD>
__device__ __forceinline__ void storeAPerm(unsigned short* rowBase, int c0, int kbe,
                                           float4 a0, float4 a1, const unsigned* sm) {
    const float v[8] = { a0.x, a0.y, a0.z, a0.w, a1.x, a1.y, a1.z, a1.w };
#pragma unroll
    for (int p = 0; p < 4; ++p) {
        const int c = c0 + 2 * p;
        const float f0 = __uint_as_float(__float_as_uint(v[2 * p])     ^ sm[2 * p]);
        const float f1 = __uint_as_float(__float_as_uint(v[2 * p + 1]) ^ sm[2 * p + 1]);
        const unsigned pk = KODD ? pack_bf16(f1, f0) : pack_bf16(f0, f1);
        *(unsigned*)(rowBase + (c ^ kbe)) = pk;
    }
}
__device__ __forceinline__ void storeBcvt(unsigned short* dst,
                                          float4 b0, float4 b1, float4 b2, float4 b3) {
    uint4 q0, q1;
    q0.x = pack_bf16(b0.x, b0.y);  q0.y = pack_bf16(b0.z, b0.w);
    q0.z = pack_bf16(b1.x, b1.y);  q0.w = pack_bf16(b1.z, b1.w);
    q1.x = pack_bf16(b2.x, b2.y);  q1.y = pack_bf16(b2.z, b2.w);
    q1.z = pack_bf16(b3.x, b3.y);  q1.w = pack_bf16(b3.z, b3.w);
    ((uint4*)dst)[0] = q0;
    ((uint4*)dst)[1] = q1;
}

template <bool KODD>
__device__ __forceinline__
void ga_gemm_fused_body(const float* __restrict__ x, const float* __restrict__ w,
                        const float* __restrict__ bias, float* __restrict__ out) {
    __shared__ __align__(16) unsigned short lA[2][TM * TK];
    __shared__ __align__(16) unsigned short lB[2][TN * TK];

    const int tid   = threadIdx.x;
    const int uBase = blockIdx.x * TN;
    const int mBase = blockIdx.y * TM;
    const int kb    = blockIdx.z;
    const int kbe   = kb & 14;

    const int aRow = tid >> 2;
    const int q    = tid & 3;
    const int ah   = q >> 1;
    const int c0   = (q & 1) << 3;
    const float* aPtr = x + (size_t)(mBase + aRow) * NI + (size_t)ah * 16 + c0;

    unsigned sgn = 0;
#pragma unroll
    for (int c = 0; c < 16; ++c)
        sgn |= (unsigned)ga_swaps_odd(c, c ^ kb) << c;
    unsigned sm[8];
#pragma unroll
    for (int e = 0; e < 8; ++e)
        sm[e] = ((sgn >> (c0 + e)) & 1u) << 31;

    const int bRow  = tid >> 1;
    const int bHalf = tid & 1;
    const float* bPtr = w + (size_t)(uBase + bRow) * NI + (size_t)bHalf * 16;

    const int wave = tid >> 5;
    const int wr   = wave >> 2;
    const int wc   = wave & 3;
    const int lane = tid & 31;
    const int r16  = lane & 15;
    const int hh   = lane >> 4;

    v8f acc[2][2] = {};

    {
        const float4* ap = (const float4*)aPtr;
        float4 a0 = ap[0], a1 = ap[1];
        const float4* bp = (const float4*)bPtr;
        float4 b0 = bp[0], b1 = bp[1], b2 = bp[2], b3 = bp[3];
        storeAPerm<KODD>(&lA[0][aRow * TK + ah * 16], c0, kbe, a0, a1, sm);
        storeBcvt(&lB[0][bRow * TK + bHalf * 16], b0, b1, b2, b3);
    }
    __syncthreads();

    for (int kt = 0; kt < NKT; ++kt) {
        const int cur = kt & 1, nxt = cur ^ 1;
        float4 a0, a1, b0, b1, b2, b3;
        const bool pre = (kt + 1 < NKT);
        if (pre) {
            const float4* ap = (const float4*)(aPtr + (size_t)(kt + 1) * TK);
            a0 = ap[0]; a1 = ap[1];
            const float4* bp = (const float4*)(bPtr + (size_t)(kt + 1) * TK);
            b0 = bp[0]; b1 = bp[1]; b2 = bp[2]; b3 = bp[3];
        }
        v16bf fa0 = loadFragA(lA[cur], wr * 32 + 0 * 16 + r16, hh);
        v16bf fa1 = loadFragA(lA[cur], wr * 32 + 1 * 16 + r16, hh);
        v16bf fb0 = loadFragB(lB[cur], wc * 32 + 0 * 16 + r16, hh);
        v16bf fb1 = loadFragB(lB[cur], wc * 32 + 1 * 16 + r16, hh);
        acc[0][0] = __builtin_amdgcn_wmma_f32_16x16x32_bf16(false, fa0, false, fb0,
                        (short)0, acc[0][0], false, false);
        acc[0][1] = __builtin_amdgcn_wmma_f32_16x16x32_bf16(false, fa0, false, fb1,
                        (short)0, acc[0][1], false, false);
        acc[1][0] = __builtin_amdgcn_wmma_f32_16x16x32_bf16(false, fa1, false, fb0,
                        (short)0, acc[1][0], false, false);
        acc[1][1] = __builtin_amdgcn_wmma_f32_16x16x32_bf16(false, fa1, false, fb1,
                        (short)0, acc[1][1], false, false);
        if (pre) {
            storeAPerm<KODD>(&lA[nxt][aRow * TK + ah * 16], c0, kbe, a0, a1, sm);
            storeBcvt(&lB[nxt][bRow * TK + bHalf * 16], b0, b1, b2, b3);
        }
        __syncthreads();
    }

#pragma unroll
    for (int nj = 0; nj < 2; ++nj) {
        const int ucol = uBase + wc * 32 + nj * 16 + r16;
        const float bv = bias[ucol * 16 + kb];
#pragma unroll
        for (int mi = 0; mi < 2; ++mi) {
            AccU a; a.v = acc[mi][nj];
            const int rowBase = mBase + wr * 32 + mi * 16 + hh * 8;
#pragma unroll
            for (int e = 0; e < 8; ++e)
                out[(size_t)(rowBase + e) * OUTC + ucol * 16 + kb] = a.f[e] + bv;
        }
    }
}

__global__ __launch_bounds__(256)
void ga_gemm_fused(const float* __restrict__ x, const float* __restrict__ w,
                   const float* __restrict__ bias, float* __restrict__ out) {
    if (blockIdx.z & 1) ga_gemm_fused_body<true>(x, w, bias, out);
    else                ga_gemm_fused_body<false>(x, w, bias, out);
}

// ================================ launch =====================================
extern "C" void kernel_launch(void* const* d_in, const int* in_sizes, int n_in,
                              void* d_out, int out_size, void* d_ws, size_t ws_size,
                              hipStream_t stream) {
    (void)in_sizes; (void)n_in; (void)out_size;
    const float* x    = (const float*)d_in[0];
    const float* w    = (const float*)d_in[1];
    const float* bias = (const float*)d_in[2];
    float* out = (float*)d_out;

    const dim3 gemmGrid(UNITS / TN, B_DIM / TM, 16);   // (4, 2, 16) = 128 workgroups
    const size_t wsNeed = (WELEMS + XELEMS) * sizeof(unsigned short);   // 10 MB

    if (d_ws != nullptr && ws_size >= wsNeed) {
        unsigned short* wsW = (unsigned short*)d_ws;
        unsigned short* wsX = wsW + WELEMS;
        const int cvtBlocks = (int)((WELEMS + XELEMS) / (8 * 256));     // 2560
        cvt_to_bf16<<<cvtBlocks, 256, 0, stream>>>(w, x, wsW, wsX);
        ga_gemm_ws<<<gemmGrid, dim3(256, 1, 1), 0, stream>>>(wsX, wsW, bias, out);
    } else {
        ga_gemm_fused<<<gemmGrid, dim3(256, 1, 1), 0, stream>>>(x, w, bias, out);
    }
}
